// MPModel_6691559047235
// MI455X (gfx1250) — matrix operand order repaired and checked by reading, not compiled, not run on hardware
//
#include <hip/hip_runtime.h>
#include <stdint.h>

#define NN 50000
#define EE 800000
#define GG 256
#define HH 128
#define LL 3

typedef unsigned short u16;
typedef __bf16 v16bf __attribute__((ext_vector_type(16)));
typedef float  v8f   __attribute__((ext_vector_type(8)));

struct __align__(16) U128 { unsigned x, y, z, w; };

union BF16x16 {
    v16bf v;
    U128  q[2];
    u16   s[16];
};

// Native f32 -> bf16 convert (backend pairs these into v_cvt_pk_bf16_f32).
__device__ __forceinline__ __bf16 f2bf(float f) { return (__bf16)f; }

__device__ __forceinline__ u16 bfbits(float f) {
    union { __bf16 b; u16 s; } u;
    u.b = (__bf16)f;
    return u.s;
}

__device__ __forceinline__ v8f v8zero() {
    v8f z;
#pragma unroll
    for (int i = 0; i < 8; ++i) z[i] = 0.0f;
    return z;
}

__device__ __forceinline__ v8f wmma_bf16(v16bf a, v16bf b, v8f c) {
    // D = A(16x32 bf16) * B(32x16 bf16) + C(16x16 f32)
    return __builtin_amdgcn_wmma_f32_16x16x32_bf16(false, a, false, b, (short)0, c, false, false);
}

// A operand (16x32 bf16 tile): this lane's 16 halves from a row-major bf16 row.
// Lane l holds row (l&15); K halves = two contiguous runs of 8.
__device__ __forceinline__ v16bf load_a_bf(const u16* row, int kt, int lane) {
    const int base = kt * 32 + ((lane >> 4) << 3);
    BF16x16 r;
    r.q[0] = *(const U128*)(row + base);
    r.q[1] = *(const U128*)(row + base + 16);
    return r.v;
}

// A operand from an f32 row, converting to bf16 (native cvt), zero-padded past Kreal.
__device__ __forceinline__ v16bf load_a_f32(const float* row, int kt, int lane, int Kreal) {
    const int base = kt * 32 + ((lane >> 4) << 3);
    v16bf r;
#pragma unroll
    for (int e = 0; e < 8; ++e) {
        const int k0 = base + e;
        const int k1 = base + 16 + e;
        r[e]     = (k0 < Kreal) ? f2bf(row[k0]) : f2bf(0.0f);
        r[e + 8] = (k1 < Kreal) ? f2bf(row[k1]) : f2bf(0.0f);
    }
    return r;
}

// B operand from pre-packed buffer (global or LDS): 16 contiguous halves.
__device__ __forceinline__ v16bf load_b(const u16* packed, int numKt, int nt, int kt, int lane) {
    const u16* p = packed + ((((size_t)nt * numKt + kt) * 32 + lane) << 4);
    BF16x16 r;
    r.q[0] = ((const U128*)p)[0];
    r.q[1] = ((const U128*)p)[1];
    return r.v;
}

// ---------------- weight packing ----------------
// src: [Kreal, HH] row-major f32.  dst layout: [nt][kt][lane][16 halves] bf16,
// where B element (k, col): k = kt*32 + ((lane>=16)?8:0) + (e&7) + ((e>=8)?16:0), col = nt*16 + (lane&15).
__global__ void pack_w_k(const float* __restrict__ src, u16* __restrict__ dst,
                         int Kreal, int Kp) {
    const int idx = blockIdx.x * 256 + threadIdx.x;
    const int numKt = Kp >> 5;
    const int total = 8 * numKt * 512;
    if (idx >= total) return;
    const int e  = idx & 15;
    const int l  = (idx >> 4) & 31;
    const int t  = idx >> 9;
    const int kt = t % numKt;
    const int nt = t / numKt;
    const int k   = kt * 32 + ((l >= 16) ? 8 : 0) + (e & 7) + ((e >= 8) ? 16 : 0);
    const int col = nt * 16 + (l & 15);
    const float v = (k < Kreal) ? src[(size_t)k * HH + col] : 0.0f;
    dst[idx] = bfbits(v);
}

__global__ void zero_f32_k(float* __restrict__ p, long n) {
    const long i = (long)blockIdx.x * 256 + threadIdx.x;
    if (i < n) p[i] = 0.0f;
}

// ---------------- node encoder: h = LN(x @ W + b) ----------------
__global__ __launch_bounds__(128, 1) void encode_nodes_k(
    const float* __restrict__ x, const u16* __restrict__ Wp,
    const float* __restrict__ bias, const float* __restrict__ lng,
    const float* __restrict__ lnb, float* __restrict__ h_f32,
    u16* __restrict__ h_bf)
{
    const int lane = threadIdx.x & 31;
    const int wave = threadIdx.x >> 5;
    const int tile = blockIdx.x * 4 + wave;
    const bool active = tile < (NN / 16);
    __shared__ __align__(16) float lbuf[4][16 * HH];
    __shared__ float lstat[4][32];
    const int row0 = tile * 16;
    v8f c[8];
#pragma unroll
    for (int nt = 0; nt < 8; ++nt) c[nt] = v8zero();
    if (active) {
        const float* arow = x + (size_t)(row0 + (lane & 15)) * 64;
#pragma unroll
        for (int kt = 0; kt < 2; ++kt) {
            v16bf a = load_a_f32(arow, kt, lane, 64);
#pragma unroll
            for (int nt = 0; nt < 8; ++nt)
                c[nt] = wmma_bf16(a, load_b(Wp, 2, nt, kt, lane), c[nt]);
        }
        const int colb = lane & 15;
        const int rb = (lane >> 4) * 8;
#pragma unroll
        for (int nt = 0; nt < 8; ++nt) {
            const int col = nt * 16 + colb;
            const float bv = bias[col];
#pragma unroll
            for (int r = 0; r < 8; ++r)
                lbuf[wave][(rb + r) * HH + col] = c[nt][r] + bv;
        }
    }
    __syncthreads();
    if (active && lane < 16) {
        const float* p = &lbuf[wave][lane * HH];
        float s = 0.f, s2 = 0.f;
#pragma unroll 8
        for (int i = 0; i < HH; ++i) { const float v = p[i]; s += v; s2 += v * v; }
        const float m = s * (1.0f / HH);
        const float var = s2 * (1.0f / HH) - m * m;
        lstat[wave][lane] = m;
        lstat[wave][16 + lane] = rsqrtf(var + 1e-5f);
    }
    __syncthreads();
    if (active) {
        for (int i = lane; i < 16 * HH; i += 32) {
            const int row = i >> 7, col = i & 127;
            const float v = (lbuf[wave][i] - lstat[wave][row]) * lstat[wave][16 + row]
                            * lng[col] + lnb[col];
            const size_t gi = (size_t)(row0 + row) * HH + col;
            h_f32[gi] = v;
            h_bf[gi]  = bfbits(v);
        }
    }
}

// ---------------- edge encoder: e = LN(edge_attr @ W + b), bf16 out ----------------
__global__ __launch_bounds__(128, 1) void encode_edges_k(
    const float* __restrict__ ea, const u16* __restrict__ Wp,
    const float* __restrict__ bias, const float* __restrict__ lng,
    const float* __restrict__ lnb, u16* __restrict__ e_bf)
{
    const int lane = threadIdx.x & 31;
    const int wave = threadIdx.x >> 5;
    const int tile = blockIdx.x * 4 + wave;       // EE/16 tiles, exact grid
    __shared__ __align__(16) float lbuf[4][16 * HH];
    __shared__ float lstat[4][32];
    const int row0 = tile * 16;
    v8f c[8];
#pragma unroll
    for (int nt = 0; nt < 8; ++nt) c[nt] = v8zero();
    {
        const float* arow = ea + (size_t)(row0 + (lane & 15)) * 16;
        v16bf a = load_a_f32(arow, 0, lane, 16);   // K=16 padded to 32
#pragma unroll
        for (int nt = 0; nt < 8; ++nt)
            c[nt] = wmma_bf16(a, load_b(Wp, 1, nt, 0, lane), c[nt]);
        const int colb = lane & 15;
        const int rb = (lane >> 4) * 8;
#pragma unroll
        for (int nt = 0; nt < 8; ++nt) {
            const int col = nt * 16 + colb;
            const float bv = bias[col];
#pragma unroll
            for (int r = 0; r < 8; ++r)
                lbuf[wave][(rb + r) * HH + col] = c[nt][r] + bv;
        }
    }
    __syncthreads();
    if (lane < 16) {
        const float* p = &lbuf[wave][lane * HH];
        float s = 0.f, s2 = 0.f;
#pragma unroll 8
        for (int i = 0; i < HH; ++i) { const float v = p[i]; s += v; s2 += v * v; }
        const float m = s * (1.0f / HH);
        const float var = s2 * (1.0f / HH) - m * m;
        lstat[wave][lane] = m;
        lstat[wave][16 + lane] = rsqrtf(var + 1e-5f);
    }
    __syncthreads();
    for (int i = lane; i < 16 * HH; i += 32) {
        const int row = i >> 7, col = i & 127;
        const float v = (lbuf[wave][i] - lstat[wave][row]) * lstat[wave][16 + row]
                        * lng[col] + lnb[col];
        e_bf[(size_t)(row0 + row) * HH + col] = bfbits(v);
    }
}

// ---------------- edge layer: gate/msg MLPs + scatter-add ----------------
// 256 threads = 8 waves; all layer weights staged in dynamic LDS (B-operand
// layout) so WMMA B operands come from ds_load_b128; 16 edge tiles per block.
#define EW 8            // waves per block
#define ER 2            // tile rounds per block
#define EDGE_TPB (EW * ER)                         // 16 tiles / block
#define EDGE_SMEM_U16 (81920 + EW * 16 * HH)       // weights + per-wave tbuf

__global__ __launch_bounds__(256, 1) void edge_layer_k(
    const u16* __restrict__ h_bf, const u16* __restrict__ e_bf,
    const int* __restrict__ src, const int* __restrict__ dst,
    const u16* __restrict__ Wg, const float* __restrict__ bg,
    const u16* __restrict__ W1, const float* __restrict__ b1,
    const u16* __restrict__ W2, const float* __restrict__ b2,
    float* __restrict__ agg)
{
    extern __shared__ u16 smem[];
    u16* swg = smem;            // 32768 halves (gate W, K=256)
    u16* sw1 = smem + 32768;    // 32768 halves (msg W1, K=256)
    u16* sw2 = smem + 65536;    // 16384 halves (msg W2, K=128)
    u16* stb = smem + 81920;    // EW * 2048 halves (relu(t) staging)
    const int tid  = threadIdx.x;
    const int lane = tid & 31;
    const int wave = tid >> 5;

    // Stage all weights for this layer into LDS (128-bit copies).
    {
        const U128* gw = (const U128*)Wg;
        const U128* g1 = (const U128*)W1;
        const U128* g2 = (const U128*)W2;
        U128* s0 = (U128*)swg;
        U128* s1 = (U128*)sw1;
        U128* s2 = (U128*)sw2;
        for (int i = tid; i < 2048; i += 256) { s0[i] = gw[i]; s1[i] = g1[i]; }
        for (int i = tid; i < 1024; i += 256) { s2[i] = g2[i]; }
    }
    __syncthreads();

    const int colb = lane & 15;
    const int rb = (lane >> 4) * 8;
    u16* tbuf = stb + wave * (16 * HH);

    for (int it = 0; it < ER; ++it) {
        const int tile = blockIdx.x * EDGE_TPB + it * EW + wave;   // exact grid
        const int e0 = tile * 16;
        const int myedge = e0 + colb;
        const u16* hrow = h_bf + (size_t)src[myedge] * HH;
        const u16* erow = e_bf + (size_t)myedge * HH;
        if (it == 0) {
            // warm next round's gathered rows (global_prefetch_b8)
            const int nedge = e0 + EW * 16 + colb;
            __builtin_prefetch(h_bf + (size_t)src[nedge] * HH, 0, 3);
            __builtin_prefetch(e_bf + (size_t)nedge * HH, 0, 3);
        }

        v8f cg[8], cm[8];
#pragma unroll
        for (int nt = 0; nt < 8; ++nt) { cg[nt] = v8zero(); cm[nt] = v8zero(); }

        // cat = [h[src], e]  (K = 256): kt 0..3 -> h, kt 4..7 -> e
#pragma unroll
        for (int kt = 0; kt < 8; ++kt) {
            v16bf a = (kt < 4) ? load_a_bf(hrow, kt, lane) : load_a_bf(erow, kt - 4, lane);
#pragma unroll
            for (int nt = 0; nt < 8; ++nt) {
                cg[nt] = wmma_bf16(a, load_b(swg, 8, nt, kt, lane), cg[nt]);
                cm[nt] = wmma_bf16(a, load_b(sw1, 8, nt, kt, lane), cm[nt]);
            }
        }
        // t = relu(cat@W1 + b1) -> LDS (bf16, A-operand-friendly row-major)
#pragma unroll
        for (int nt = 0; nt < 8; ++nt) {
            const int col = nt * 16 + colb;
            const float bv = b1[col];
#pragma unroll
            for (int r = 0; r < 8; ++r) {
                float v = cm[nt][r] + bv;
                v = v > 0.f ? v : 0.f;
                tbuf[(rb + r) * HH + col] = bfbits(v);
            }
        }
        __syncthreads();
#pragma unroll
        for (int nt = 0; nt < 8; ++nt) cm[nt] = v8zero();
        const u16* trow = tbuf + colb * HH;
#pragma unroll
        for (int kt = 0; kt < 4; ++kt) {
            v16bf a = load_a_bf(trow, kt, lane);
#pragma unroll
            for (int nt = 0; nt < 8; ++nt)
                cm[nt] = wmma_bf16(a, load_b(sw2, 4, nt, kt, lane), cm[nt]);
        }
        int drow[8];
#pragma unroll
        for (int r = 0; r < 8; ++r) drow[r] = dst[e0 + rb + r];
#pragma unroll
        for (int nt = 0; nt < 8; ++nt) {
            const int col = nt * 16 + colb;
            const float gbv = bg[col];
            const float mbv = b2[col];
#pragma unroll
            for (int r = 0; r < 8; ++r) {
                const float gate = 1.0f / (1.0f + __expf(-(cg[nt][r] + gbv)));
                const float msg  = cm[nt][r] + mbv;
                atomicAdd(&agg[(size_t)drow[r] * HH + col], gate * msg);
            }
        }
        __syncthreads();
    }
}

// ---------------- node layer: residual MLP on [h, agg] + LN ----------------
// 256 threads = 8 waves; res1+res2 staged in dynamic LDS; 32 node tiles / block.
#define NW 8            // waves per block
#define NR 4            // tile rounds per block
#define NODE_TPB (NW * NR)                          // 32 tiles / block
#define NODE_TILES (NN / 16)                        // 3125
// u16 units: w1(32768) + w2(16384) + tbuf(NW*2048) + lbuf(NW*2048 f32 = NW*4096 u16)
//            + lstat(NW*32 f32 = NW*64 u16)
#define NODE_SMEM_U16 (32768 + 16384 + NW * 2048 + NW * 4096 + NW * 64)

__global__ __launch_bounds__(256, 1) void node_layer_k(
    float* __restrict__ h_f32, u16* __restrict__ h_bf,
    const float* __restrict__ agg,
    const u16* __restrict__ W1, const float* __restrict__ b1,
    const u16* __restrict__ W2, const float* __restrict__ b2,
    const float* __restrict__ lng, const float* __restrict__ lnb)
{
    extern __shared__ u16 smem[];
    u16*   sw1 = smem;                          // 32768 halves (res W1, K=256)
    u16*   sw2 = smem + 32768;                  // 16384 halves (res W2, K=128)
    u16*   stb = smem + 49152;                  // NW * 2048 halves
    float* slb = (float*)(smem + 49152 + NW * 2048);   // NW * 2048 floats
    float* sst = slb + NW * 2048;               // NW * 32 floats
    const int tid  = threadIdx.x;
    const int lane = tid & 31;
    const int wave = tid >> 5;
    {
        const U128* g1 = (const U128*)W1;
        const U128* g2 = (const U128*)W2;
        U128* s1 = (U128*)sw1;
        U128* s2 = (U128*)sw2;
        for (int i = tid; i < 2048; i += 256) s1[i] = g1[i];
        for (int i = tid; i < 1024; i += 256) s2[i] = g2[i];
    }
    __syncthreads();

    const int colb = lane & 15;
    const int rb = (lane >> 4) * 8;
    u16*   tbuf = stb + wave * (16 * HH);
    float* lbuf = slb + wave * (16 * HH);
    float* lst  = sst + wave * 32;

    for (int it = 0; it < NR; ++it) {
        const int tile = blockIdx.x * NODE_TPB + it * NW + wave;
        const bool active = tile < NODE_TILES;
        const int row0 = tile * 16;
        v8f c[8];
#pragma unroll
        for (int nt = 0; nt < 8; ++nt) c[nt] = v8zero();
        if (active) {
            const u16*   hrow = h_bf + (size_t)(row0 + colb) * HH;
            const float* arow = agg  + (size_t)(row0 + colb) * HH;
#pragma unroll
            for (int kt = 0; kt < 8; ++kt) {
                v16bf a = (kt < 4) ? load_a_bf(hrow, kt, lane)
                                   : load_a_f32(arow, kt - 4, lane, HH);
#pragma unroll
                for (int nt = 0; nt < 8; ++nt)
                    c[nt] = wmma_bf16(a, load_b(sw1, 8, nt, kt, lane), c[nt]);
            }
#pragma unroll
            for (int nt = 0; nt < 8; ++nt) {
                const int col = nt * 16 + colb;
                const float bv = b1[col];
#pragma unroll
                for (int r = 0; r < 8; ++r) {
                    float v = c[nt][r] + bv;
                    v = v > 0.f ? v : 0.f;
                    tbuf[(rb + r) * HH + col] = bfbits(v);
                }
            }
        }
        __syncthreads();
        if (active) {
#pragma unroll
            for (int nt = 0; nt < 8; ++nt) c[nt] = v8zero();
            const u16* trow = tbuf + colb * HH;
#pragma unroll
            for (int kt = 0; kt < 4; ++kt) {
                v16bf a = load_a_bf(trow, kt, lane);
#pragma unroll
                for (int nt = 0; nt < 8; ++nt)
                    c[nt] = wmma_bf16(a, load_b(sw2, 4, nt, kt, lane), c[nt]);
            }
            // pre-LN = h_old + res
#pragma unroll
            for (int nt = 0; nt < 8; ++nt) {
                const int col = nt * 16 + colb;
                const float bv = b2[col];
#pragma unroll
                for (int r = 0; r < 8; ++r) {
                    const int row = rb + r;
                    const size_t gi = (size_t)(row0 + row) * HH + col;
                    lbuf[row * HH + col] = h_f32[gi] + c[nt][r] + bv;
                }
            }
        }
        __syncthreads();
        if (active && lane < 16) {
            const float* p = lbuf + lane * HH;
            float s = 0.f, s2 = 0.f;
#pragma unroll 8
            for (int i = 0; i < HH; ++i) { const float v = p[i]; s += v; s2 += v * v; }
            const float m = s * (1.0f / HH);
            const float var = s2 * (1.0f / HH) - m * m;
            lst[lane] = m;
            lst[16 + lane] = rsqrtf(var + 1e-5f);
        }
        __syncthreads();
        if (active) {
            for (int i = lane; i < 16 * HH; i += 32) {
                const int row = i >> 7, col = i & 127;
                const float v = (lbuf[i] - lst[row]) * lst[16 + row] * lng[col] + lnb[col];
                const size_t gi = (size_t)(row0 + row) * HH + col;
                h_f32[gi] = v;
                h_bf[gi]  = bfbits(v);
            }
        }
        __syncthreads();
    }
}

// ---------------- global mean pool (atomics) ----------------
__global__ void pool_k(const float* __restrict__ h, const int* __restrict__ batch,
                       float* __restrict__ sums, float* __restrict__ cnt) {
    const long i = (long)blockIdx.x * 256 + threadIdx.x;
    if (i >= (long)NN * HH) return;
    const int n = (int)(i >> 7);
    const int c = (int)(i & 127);
    const int b = batch[n];
    atomicAdd(&sums[(size_t)b * HH + c], h[i]);
    if (c == 0) atomicAdd(&cnt[b], 1.0f);
}

// ---------------- readout MLP: [G,128] -> relu -> [G,2] ----------------
__global__ void readout_k(const float* __restrict__ sums, const float* __restrict__ cnt,
                          const float* __restrict__ w1, const float* __restrict__ b1,
                          const float* __restrict__ w2, const float* __restrict__ b2,
                          float* __restrict__ out) {
    const int g = blockIdx.x;
    const int j = threadIdx.x;
    __shared__ float gv[HH];
    __shared__ float hid[HH];
    float c = cnt[g];
    c = c > 1.0f ? c : 1.0f;
    gv[j] = sums[(size_t)g * HH + j] / c;
    __syncthreads();
    float acc = b1[j];
#pragma unroll 8
    for (int k = 0; k < HH; ++k) acc += gv[k] * w1[k * HH + j];
    hid[j] = acc > 0.f ? acc : 0.f;
    __syncthreads();
    if (j < 2) {
        float o = b2[j];
#pragma unroll 8
        for (int k = 0; k < HH; ++k) o += hid[k] * w2[k * 2 + j];
        out[g * 2 + j] = o;
    }
}

extern "C" void kernel_launch(void* const* d_in, const int* in_sizes, int n_in,
                              void* d_out, int out_size, void* d_ws, size_t ws_size,
                              hipStream_t stream) {
    (void)in_sizes; (void)n_in; (void)out_size; (void)ws_size;
    const float* x          = (const float*)d_in[0];
    const float* edge_attr  = (const float*)d_in[1];
    const int*   eidx       = (const int*)d_in[2];
    const int*   batch      = (const int*)d_in[3];
    const float* node_enc_w = (const float*)d_in[4];
    const float* node_enc_b = (const float*)d_in[5];
    const float* edge_enc_w = (const float*)d_in[6];
    const float* edge_enc_b = (const float*)d_in[7];
    const float* node_ln_g  = (const float*)d_in[8];
    const float* node_ln_b  = (const float*)d_in[9];
    const float* edge_ln_g  = (const float*)d_in[10];
    const float* edge_ln_b  = (const float*)d_in[11];
    const float* msg_w1     = (const float*)d_in[12];
    const float* msg_b1     = (const float*)d_in[13];
    const float* msg_w2     = (const float*)d_in[14];
    const float* msg_b2     = (const float*)d_in[15];
    const float* gate_w     = (const float*)d_in[16];
    const float* gate_b     = (const float*)d_in[17];
    const float* res_w1     = (const float*)d_in[18];
    const float* res_b1     = (const float*)d_in[19];
    const float* res_w2     = (const float*)d_in[20];
    const float* res_b2     = (const float*)d_in[21];
    const float* ln_g       = (const float*)d_in[22];
    const float* ln_b       = (const float*)d_in[23];
    const float* ro_w1      = (const float*)d_in[24];
    const float* ro_b1      = (const float*)d_in[25];
    const float* ro_w2      = (const float*)d_in[26];
    const float* ro_b2      = (const float*)d_in[27];
    float* out = (float*)d_out;

    char* ws = (char*)d_ws;
    size_t off = 0;
    auto alloc = [&](size_t bytes) -> void* {
        void* p = ws + off;
        off += (bytes + 255) & ~(size_t)255;
        return p;
    };
    float* h_f32 = (float*)alloc((size_t)NN * HH * 4);
    u16*   h_bf  = (u16*)  alloc((size_t)NN * HH * 2);
    u16*   e_bf  = (u16*)  alloc((size_t)EE * HH * 2);
    float* agg   = (float*)alloc((size_t)NN * HH * 4);
    float* sums  = (float*)alloc((size_t)GG * HH * 4);
    float* cnt   = (float*)alloc((size_t)GG * 4);
    u16* p_node = (u16*)alloc((size_t)8192 * 2);
    u16* p_edge = (u16*)alloc((size_t)4096 * 2);
    u16 *p_gate[LL], *p_m1[LL], *p_m2[LL], *p_r1[LL], *p_r2[LL];
    for (int l = 0; l < LL; ++l) {
        p_gate[l] = (u16*)alloc((size_t)32768 * 2);
        p_m1[l]   = (u16*)alloc((size_t)32768 * 2);
        p_m2[l]   = (u16*)alloc((size_t)16384 * 2);
        p_r1[l]   = (u16*)alloc((size_t)32768 * 2);
        p_r2[l]   = (u16*)alloc((size_t)16384 * 2);
    }

    auto pack = [&](const float* srcw, u16* dstw, int Kreal, int Kp) {
        const int total = 8 * (Kp / 32) * 512;
        pack_w_k<<<(total + 255) / 256, 256, 0, stream>>>(srcw, dstw, Kreal, Kp);
    };
    pack(node_enc_w, p_node, 64, 64);
    pack(edge_enc_w, p_edge, 16, 32);
    for (int l = 0; l < LL; ++l) {
        pack(gate_w + (size_t)l * 256 * HH, p_gate[l], 256, 256);
        pack(msg_w1 + (size_t)l * 256 * HH, p_m1[l],   256, 256);
        pack(msg_w2 + (size_t)l * 128 * HH, p_m2[l],   128, 128);
        pack(res_w1 + (size_t)l * 256 * HH, p_r1[l],   256, 256);
        pack(res_w2 + (size_t)l * 128 * HH, p_r2[l],   128, 128);
    }

    const int encNodeBlocks = (NN / 16 + 3) / 4;       // 782 (3125 tiles, guarded)
    const int encEdgeBlocks = EE / 16 / 4;              // 12500 (exact)
    const int edgeBlocks    = EE / 16 / EDGE_TPB;       // 3125 (exact)
    const int nodeBlocks    = (NODE_TILES + NODE_TPB - 1) / NODE_TPB;   // 98
    const size_t edgeSmem = (size_t)EDGE_SMEM_U16 * sizeof(u16);   // ~192 KB dynamic LDS
    const size_t nodeSmem = (size_t)NODE_SMEM_U16 * sizeof(u16);   // ~193 KB dynamic LDS

    encode_nodes_k<<<encNodeBlocks, 128, 0, stream>>>(x, p_node, node_enc_b,
                                                      node_ln_g, node_ln_b, h_f32, h_bf);
    encode_edges_k<<<encEdgeBlocks, 128, 0, stream>>>(edge_attr, p_edge, edge_enc_b,
                                                      edge_ln_g, edge_ln_b, e_bf);

    const int* srcp = eidx;
    const int* dstp = eidx + EE;
    const long aggN = (long)NN * HH;
    for (int l = 0; l < LL; ++l) {
        zero_f32_k<<<(int)((aggN + 255) / 256), 256, 0, stream>>>(agg, aggN);
        edge_layer_k<<<edgeBlocks, 256, edgeSmem, stream>>>(h_bf, e_bf, srcp, dstp,
            p_gate[l], gate_b + (size_t)l * HH,
            p_m1[l],   msg_b1 + (size_t)l * HH,
            p_m2[l],   msg_b2 + (size_t)l * HH, agg);
        node_layer_k<<<nodeBlocks, 256, nodeSmem, stream>>>(h_f32, h_bf, agg,
            p_r1[l], res_b1 + (size_t)l * HH,
            p_r2[l], res_b2 + (size_t)l * HH,
            ln_g + (size_t)l * HH, ln_b + (size_t)l * HH);
    }
    const long poolN = (long)GG * HH;
    zero_f32_k<<<(int)((poolN + 255) / 256), 256, 0, stream>>>(sums, poolN);
    zero_f32_k<<<1, 256, 0, stream>>>(cnt, GG);
    pool_k<<<(int)((aggN + 255) / 256), 256, 0, stream>>>(h_f32, batch, sums, cnt);
    readout_k<<<GG, 128, 0, stream>>>(sums, cnt, ro_w1, ro_b1, ro_w2, ro_b2, out);
}
